// Cov_Trans_Block_46961172414479
// MI455X (gfx1250) — matrix operand order repaired and checked by reading, not compiled, not run on hardware
//
#include <hip/hip_runtime.h>
#include <hip/hip_bf16.h>

typedef __attribute__((ext_vector_type(16))) _Float16 v16h;
typedef __attribute__((ext_vector_type(8)))  _Float16 v8h;
typedef __attribute__((ext_vector_type(8)))  float    v8f;
typedef __attribute__((ext_vector_type(4)))  unsigned int v4u;
typedef __attribute__((ext_vector_type(8)))  unsigned int v8u;

static __device__ __forceinline__ v8f wmma_f16(v16h a, v16h b, v8f c) {
  // D = A(16x32 f16) * B(32x16 f16) + C(16x16 f32)
  return __builtin_amdgcn_wmma_f32_16x16x32_f16(false, a, false, b, (short)0, c, false, false);
}

// 16 contiguous halves (32B) from LDS -> full v16h (2x ds_load_b128)
static __device__ __forceinline__ v16h ld16(const _Float16* p) {
  return *(const v16h*)p;
}
// two contiguous 8-half runs -> v16h
static __device__ __forceinline__ v16h ld8x2(const _Float16* p0, const _Float16* p1) {
  v8h a = *(const v8h*)p0;
  v8h b = *(const v8h*)p1;
  return __builtin_shufflevector(a, b, 0, 1, 2, 3, 4, 5, 6, 7, 8, 9, 10, 11, 12, 13, 14, 15);
}

// ---------------------------------------------------------------------------
// Tensor Data Mover: async 2-D f16 tile load Global -> LDS.
// D# per CDNA5 ISA ch.8: group0 {count=1 | lds_addr | global_addr | type=2},
// group1 {data_size=1(2B) | tensor dims | tile_d0/d1 | dim0 stride}.
// Issued wave-uniform; completion via s_wait_tensorcnt + workgroup barrier.
// ---------------------------------------------------------------------------
static __device__ __forceinline__ void tdm_load_2d_f16(unsigned lds_off,
                                                       const _Float16* gptr,
                                                       unsigned tile_d0, unsigned tile_d1,
                                                       unsigned stride_elems) {
  unsigned long long ga = (unsigned long long)(const void*)gptr;   // byte address
  v4u g0;
  g0.x = 1u;                                          // count=1, user descriptor
  g0.y = lds_off;                                     // lds_addr (bytes)
  g0.z = (unsigned)ga;                                // global_addr[31:0]
  g0.w = ((unsigned)(ga >> 32) & 0x01FFFFFFu) | (2u << 30);  // addr[56:32] | type=2
  const unsigned dim0 = 1u << 20, dim1 = 1u << 20;    // tiles are always in-bounds
  v8u g1;
  g1[0] = (1u << 16);                                 // wg_mask=0, data_size=1 (2 bytes)
  g1[1] = (dim0 & 0xFFFFu) << 16;                     // tensor_dim0[15:0]
  g1[2] = (dim0 >> 16) | ((dim1 & 0xFFFFu) << 16);    // dim0[31:16] | dim1[15:0]
  g1[3] = (dim1 >> 16) | (tile_d0 << 16);             // dim1[31:16] | tile_dim0
  g1[4] = tile_d1;                                    // tile_dim1 | tile_dim2=0
  g1[5] = stride_elems;                               // tensor_dim0_stride[31:0]
  g1[6] = 0u;                                         // stride hi | dim1_stride lo
  g1[7] = 0u;
  asm volatile("tensor_load_to_lds %0, %1" :: "s"(g0), "s"(g1) : "memory");
}

// ---------------------------------------------------------------------------
// Weight convert f32 -> f16 (layout preserved: [O][I] == [N][K], K-innermost)
// ---------------------------------------------------------------------------
__global__ __launch_bounds__(256)
void convert_f16_kernel(const float* __restrict__ src, _Float16* __restrict__ dst, int n) {
  int t = blockIdx.x * 256 + threadIdx.x;
  if (t < n) dst[t] = (_Float16)src[t];
}

// ---------------------------------------------------------------------------
// Depthwise 3x3 conv + bias + BN (inference) -> f16, layout [3][B][S][C]
// ---------------------------------------------------------------------------
__global__ __launch_bounds__(256)
void dwconv_bn_kernel(const float* __restrict__ x, const float* __restrict__ dww,
                      const float* __restrict__ dwb, const float* __restrict__ bng,
                      const float* __restrict__ bnb, const float* __restrict__ bnm,
                      const float* __restrict__ bnv, _Float16* __restrict__ out) {
  size_t t = (size_t)blockIdx.x * 256 + threadIdx.x;  // 3*4*2304*256 total
  int c    = (int)(t & 255);
  int sp   = (int)((t >> 8) % 2304);
  int rest = (int)((t >> 8) / 2304);   // i*4 + b
  int b = rest & 3;
  int i = rest >> 2;
  int yy = sp / 48, xx = sp % 48;
  const float* wb = dww + ((size_t)i * 256 + c) * 9;
  float sum = 0.f;
  #pragma unroll
  for (int ky = 0; ky < 3; ++ky) {
    int y2 = yy + ky - 1;
    if ((unsigned)y2 >= 48u) continue;
    #pragma unroll
    for (int kx = 0; kx < 3; ++kx) {
      int x2 = xx + kx - 1;
      if ((unsigned)x2 >= 48u) continue;
      sum += x[((size_t)b * 2304 + y2 * 48 + x2) * 256 + c] * wb[ky * 3 + kx];
    }
  }
  int ic = i * 256 + c;
  float scale = bng[ic] * rsqrtf(bnv[ic] + 1e-5f);
  float val = (sum + dwb[ic] - bnm[ic]) * scale + bnb[ic];
  out[t] = (_Float16)val;
}

// ---------------------------------------------------------------------------
// WMMA GEMM: D[M,N] = A[M,K](f16,row) * W[N,K](f16,row, K-innermost) + bias.
// 64x64 block, 8 waves (wave = 16x32 via two 16x16 accumulators).
// TDM stages A tile [64m][32k] and B tile [64n][32k] into LDS.
// mode 0: f16 row-major | mode 1: f16 transposed [(b*256+n)*2304+s]
// mode 2: f32 + residual | mode 3: f16 exact-GELU
// ---------------------------------------------------------------------------
__global__ __launch_bounds__(256)
void gemm_wmma_kernel(const _Float16* __restrict__ A, const _Float16* __restrict__ Bm,
                      const float* __restrict__ bias,
                      _Float16* __restrict__ outH, float* __restrict__ outF,
                      const float* __restrict__ resid,
                      int M, int N, int K, int mode) {
  __shared__ alignas(32) _Float16 sA[64 * 32];   // [m][k]
  __shared__ alignas(32) _Float16 sB[64 * 32];   // [n][k]
  const int tid = threadIdx.x, lane = tid & 31, wave = tid >> 5;
  const int laneHi = lane >> 4, lr = lane & 15;
  const int mw = wave >> 1, nw = wave & 1;
  const int blockM = blockIdx.x * 64, blockN = blockIdx.y * 64;
  const unsigned ldsA = (unsigned)(size_t)(void*)sA;
  const unsigned ldsB = (unsigned)(size_t)(void*)sB;

  v8f acc0 = {}, acc1 = {};

  for (int k0 = 0; k0 < K; k0 += 32) {
    __syncthreads();
    if (wave == 0) {
      tdm_load_2d_f16(ldsA, A  + (size_t)blockM * K + k0, 32, 64, (unsigned)K);
      tdm_load_2d_f16(ldsB, Bm + (size_t)blockN * K + k0, 32, 64, (unsigned)K);
      __builtin_amdgcn_s_wait_tensorcnt(0);
    }
    if (k0 + 32 < K)
      __builtin_prefetch(A + (size_t)(blockM + (tid & 63)) * K + k0 + 32, 0, 0);
    __syncthreads();

    const int row = mw * 16 + lr;
    v16h a  = ld16(&sA[row * 32 + laneHi * 16]);
    v16h b0 = ld16(&sB[(nw * 32 + lr) * 32 + laneHi * 16]);
    v16h b1 = ld16(&sB[(nw * 32 + 16 + lr) * 32 + laneHi * 16]);
    acc0 = wmma_f16(a, b0, acc0);
    acc1 = wmma_f16(a, b1, acc1);
  }

  #pragma unroll
  for (int j = 0; j < 8; ++j) {
    int m  = blockM + mw * 16 + j + 8 * laneHi;
    int n0 = blockN + nw * 32 + lr;
    int n1 = n0 + 16;
    float v0 = acc0[j] + bias[n0];
    float v1 = acc1[j] + bias[n1];
    if (mode == 2) {
      size_t i0 = (size_t)m * N + n0, i1 = (size_t)m * N + n1;
      outF[i0] = v0 + resid[i0];
      outF[i1] = v1 + resid[i1];
    } else if (mode == 3) {
      v0 = 0.5f * v0 * (1.f + erff(v0 * 0.70710678118f));
      v1 = 0.5f * v1 * (1.f + erff(v1 * 0.70710678118f));
      outH[(size_t)m * N + n0] = (_Float16)v0;
      outH[(size_t)m * N + n1] = (_Float16)v1;
    } else if (mode == 1) {
      int bb = m / 2304, s = m % 2304;   // N == 256 in this mode
      outH[((size_t)bb * 256 + n0) * 2304 + s] = (_Float16)v0;
      outH[((size_t)bb * 256 + n1) * 2304 + s] = (_Float16)v1;
    } else {
      outH[(size_t)m * N + n0] = (_Float16)v0;
      outH[(size_t)m * N + n1] = (_Float16)v1;
    }
  }
}

// ---------------------------------------------------------------------------
// Flash attention. grid = (18 qblocks, 4 heads, 4 batch), 256 threads.
// ktb/vtb are [B][C][S]; read row-major as [B][S][C] they ARE the reference's
// scrambled k-view (and the matching V layout). TDM stages:
//   sK2 [32 key][64 e]  (rows of the [S][C] view, stride 256)
//   sVT [64 e][32 key]  (rows of the [C][S] layout, stride 2304)
// Online softmax per 16-row wave tile over 2304 keys in 32-key chunks.
// ---------------------------------------------------------------------------
__global__ __launch_bounds__(256)
void attention_kernel(const _Float16* __restrict__ q, const _Float16* __restrict__ kt,
                      const _Float16* __restrict__ vt, _Float16* __restrict__ outp) {
  const int S = 2304;
  __shared__ alignas(32) _Float16 sK2[32 * 64];
  __shared__ alignas(32) _Float16 sVT[64 * 32];
  __shared__ float    sL[8][16 * 32];
  __shared__ alignas(32) _Float16 sP[8][16 * 32];
  __shared__ float    sCorr[8][16];
  __shared__ float    sSum[8][16];

  const int tid = threadIdx.x, lane = tid & 31, wave = tid >> 5;
  const int laneHi = lane >> 4, lr = lane & 15;
  const int h = blockIdx.y, b = blockIdx.z;
  const int qRow0 = blockIdx.x * 128 + wave * 16;
  const unsigned ldsK = (unsigned)(size_t)(void*)sK2;
  const unsigned ldsV = (unsigned)(size_t)(void*)sVT;

  // Q fragments (A-operand): one contiguous 16-half run per 16x32 chunk
  const _Float16* qp = q + ((size_t)b * S + qRow0 + lr) * 256 + h * 64;
  v16h aq0 = ld16(qp + laneHi * 16);
  v16h aq1 = ld16(qp + 32 + laneHi * 16);

  v8f o0 = {}, o1 = {}, o2 = {}, o3 = {};
  float rm = -3.0e38f, rs = 0.f;   // row max / row sum (lanes 0..15)

  for (int kc = 0; kc < S; kc += 32) {
    __syncthreads();
    if (wave == 0) {
      tdm_load_2d_f16(ldsK, kt + (size_t)b * 256 * S + (size_t)kc * 256 + h * 64,
                      64, 32, 256);
      tdm_load_2d_f16(ldsV, vt + ((size_t)b * 256 + h * 64) * S + kc,
                      32, 64, (unsigned)S);
      __builtin_amdgcn_s_wait_tensorcnt(0);
    }
    __syncthreads();

    // logits 16x32 (two N-subtiles, contraction over e=64)
    v8f L0 = {}, L1 = {};
    #pragma unroll
    for (int hf = 0; hf < 2; ++hf) {
      const v16h& aq = hf ? aq1 : aq0;
      const int e0 = hf * 32;
      v16h bk0 = ld16(&sK2[lr * 64        + e0 + laneHi * 16]);
      v16h bk1 = ld16(&sK2[(16 + lr) * 64 + e0 + laneHi * 16]);
      L0 = wmma_f16(aq, bk0, L0);
      L1 = wmma_f16(aq, bk1, L1);
    }
    #pragma unroll
    for (int j = 0; j < 8; ++j) {
      int row = j + 8 * laneHi;
      sL[wave][row * 32 + lr]      = L0[j] * 0.125f;
      sL[wave][row * 32 + 16 + lr] = L1[j] * 0.125f;
    }
    __syncthreads();

    // online softmax: lane r (0..15) owns query row r
    if (lane < 16) {
      float* Lr = &sL[wave][lane * 32];
      float mx = rm;
      #pragma unroll
      for (int i2 = 0; i2 < 32; ++i2) mx = fmaxf(mx, Lr[i2]);
      float corr = __expf(rm - mx);
      float ssum = 0.f;
      _Float16* Pr = &sP[wave][lane * 32];
      #pragma unroll
      for (int i2 = 0; i2 < 32; ++i2) {
        float pe = __expf(Lr[i2] - mx);
        Pr[i2] = (_Float16)pe;
        ssum += pe;
      }
      rs = rs * corr + ssum;
      rm = mx;
      sCorr[wave][lane] = corr;
    }
    __syncthreads();

    // rescale accumulators, then O += P(16x32) * V(32x64)
    float cr[8];
    #pragma unroll
    for (int j = 0; j < 8; ++j) cr[j] = sCorr[wave][j + 8 * laneHi];
    #pragma unroll
    for (int j = 0; j < 8; ++j) { o0[j] *= cr[j]; o1[j] *= cr[j]; o2[j] *= cr[j]; o3[j] *= cr[j]; }

    v16h ap = ld16(&sP[wave][lr * 32 + laneHi * 16]);
    #pragma unroll
    for (int t = 0; t < 4; ++t) {
      v16h bv = ld16(&sVT[(t * 16 + lr) * 32 + laneHi * 16]);
      v8f& ot = (t == 0) ? o0 : (t == 1) ? o1 : (t == 2) ? o2 : o3;
      ot = wmma_f16(ap, bv, ot);
    }
  }

  if (lane < 16) sSum[wave][lane] = rs;
  __syncthreads();
  float invs[8];
  #pragma unroll
  for (int j = 0; j < 8; ++j) invs[j] = 1.f / sSum[wave][j + 8 * laneHi];
  const size_t obase = ((size_t)b * S + qRow0) * 256 + h * 64;
  #pragma unroll
  for (int j = 0; j < 8; ++j) {
    size_t ro = obase + (size_t)(j + 8 * laneHi) * 256;
    outp[ro + lr]      = (_Float16)(o0[j] * invs[j]);
    outp[ro + 16 + lr] = (_Float16)(o1[j] * invs[j]);
    outp[ro + 32 + lr] = (_Float16)(o2[j] * invs[j]);
    outp[ro + 48 + lr] = (_Float16)(o3[j] * invs[j]);
  }
}

// ---------------------------------------------------------------------------
// LayerNorm over C=256: one wave per row, 8 elems/lane, shfl reductions -> f16
// ---------------------------------------------------------------------------
__global__ __launch_bounds__(256)
void layernorm_kernel(const float* __restrict__ in, const float* __restrict__ g,
                      const float* __restrict__ bta, _Float16* __restrict__ out) {
  int wave = threadIdx.x >> 5, lane = threadIdx.x & 31;
  int row = blockIdx.x * 8 + wave;
  const float* p = in + (size_t)row * 256;
  float vals[8], s = 0.f, s2 = 0.f;
  #pragma unroll
  for (int j = 0; j < 8; ++j) { float v = p[lane * 8 + j]; vals[j] = v; s += v; s2 += v * v; }
  #pragma unroll
  for (int o = 16; o > 0; o >>= 1) { s += __shfl_xor(s, o, 32); s2 += __shfl_xor(s2, o, 32); }
  float mean = s * (1.f / 256.f);
  float var  = s2 * (1.f / 256.f) - mean * mean;
  float rstd = rsqrtf(var + 1e-5f);
  #pragma unroll
  for (int j = 0; j < 8; ++j) {
    int c = lane * 8 + j;
    out[(size_t)row * 256 + c] = (_Float16)((vals[j] - mean) * rstd * g[c] + bta[c]);
  }
}

// ---------------------------------------------------------------------------
extern "C" void kernel_launch(void* const* d_in, const int* in_sizes, int n_in,
                              void* d_out, int out_size, void* d_ws, size_t ws_size,
                              hipStream_t stream) {
  (void)in_sizes; (void)n_in; (void)out_size; (void)ws_size;
  const float* x    = (const float*)d_in[0];
  const float* dw_w = (const float*)d_in[3];
  const float* dw_b = (const float*)d_in[4];
  const float* bn_g = (const float*)d_in[5];
  const float* bn_b = (const float*)d_in[6];
  const float* bn_m = (const float*)d_in[7];
  const float* bn_v = (const float*)d_in[8];
  const float* pw_w = (const float*)d_in[9];
  const float* pw_b = (const float*)d_in[10];
  const float* wo_w = (const float*)d_in[11];
  const float* wo_b = (const float*)d_in[12];
  const float* ln_g = (const float*)d_in[13];
  const float* ln_b = (const float*)d_in[14];
  const float* w1_w = (const float*)d_in[15];
  const float* w1_b = (const float*)d_in[16];
  const float* w2_w = (const float*)d_in[17];
  const float* w2_b = (const float*)d_in[18];
  float* out = (float*)d_out;

  char* ws = (char*)d_ws;
  size_t off = 0;
  auto take = [&](size_t bytes) -> char* {
    char* p = ws + off;
    off += (bytes + 255) & ~(size_t)255;
    return p;
  };
  _Float16* pwc  = (_Float16*)take((size_t)3 * 256 * 256 * 2);   // [3][O][I]
  _Float16* woc  = (_Float16*)take((size_t)256 * 256 * 2);       // [O][I]
  _Float16* w1c  = (_Float16*)take((size_t)1024 * 256 * 2);      // [O][I]
  _Float16* w2c  = (_Float16*)take((size_t)256 * 1024 * 2);      // [O][I]
  _Float16* dwh  = (_Float16*)take((size_t)3 * 9216 * 256 * 2);
  _Float16* qh   = (_Float16*)take((size_t)9216 * 256 * 2);
  _Float16* ktb  = (_Float16*)take((size_t)9216 * 256 * 2);      // [B][C][S]
  _Float16* vtb  = (_Float16*)take((size_t)9216 * 256 * 2);      // [B][C][S]
  _Float16* ath  = (_Float16*)take((size_t)9216 * 256 * 2);
  float*    out1 = (float*)   take((size_t)9216 * 256 * 4);
  _Float16* xnh  = (_Float16*)take((size_t)9216 * 256 * 2);
  _Float16* hddh = (_Float16*)take((size_t)9216 * 1024 * 2);

  convert_f16_kernel<<<(3 * 65536) / 256, 256, 0, stream>>>(pw_w, pwc, 3 * 65536);
  convert_f16_kernel<<<65536 / 256, 256, 0, stream>>>(wo_w, woc, 65536);
  convert_f16_kernel<<<262144 / 256, 256, 0, stream>>>(w1_w, w1c, 262144);
  convert_f16_kernel<<<262144 / 256, 256, 0, stream>>>(w2_w, w2c, 262144);

  dwconv_bn_kernel<<<(3 * 9216 * 256) / 256, 256, 0, stream>>>(
      x, dw_w, dw_b, bn_g, bn_b, bn_m, bn_v, dwh);

  dim3 g256(144, 4), g1024(144, 16);
  // q (row-major) / k,v (transposed [B][C][S] stores) projections
  gemm_wmma_kernel<<<g256, 256, 0, stream>>>(dwh,                          pwc,          pw_b,       qh,  nullptr, nullptr, 9216, 256, 256, 0);
  gemm_wmma_kernel<<<g256, 256, 0, stream>>>(dwh + (size_t)9216 * 256,     pwc + 65536,  pw_b + 256, ktb, nullptr, nullptr, 9216, 256, 256, 1);
  gemm_wmma_kernel<<<g256, 256, 0, stream>>>(dwh + (size_t)2 * 9216 * 256, pwc + 131072, pw_b + 512, vtb, nullptr, nullptr, 9216, 256, 256, 1);

  attention_kernel<<<dim3(18, 4, 4), 256, 0, stream>>>(qh, ktb, vtb, ath);

  // wo projection + residual(x) -> out1 (f32)
  gemm_wmma_kernel<<<g256, 256, 0, stream>>>(ath, woc, wo_b, nullptr, out1, x, 9216, 256, 256, 2);

  layernorm_kernel<<<9216 / 8, 256, 0, stream>>>(out1, ln_g, ln_b, xnh);

  // FFN
  gemm_wmma_kernel<<<g1024, 256, 0, stream>>>(xnh, w1c, w1_b, hddh, nullptr, nullptr, 9216, 1024, 256, 3);
  gemm_wmma_kernel<<<g256, 256, 0, stream>>>(hddh, w2c, w2_b, nullptr, out, out1, 9216, 256, 1024, 2);
}
